// MultiHeadEncoder_87205015978171
// MI455X (gfx1250) — compile-verified, matrix operand order checked by reading, not run on hardware
//
#include <hip/hip_runtime.h>
#include <hip/hip_bf16.h>

// ---------------------------------------------------------------------------
// Types
// ---------------------------------------------------------------------------
typedef __attribute__((ext_vector_type(16))) _Float16 v16h;
typedef __attribute__((ext_vector_type(8)))  _Float16 v8h;
typedef __attribute__((ext_vector_type(2)))  _Float16 h2;
typedef __attribute__((ext_vector_type(8)))  float    v8f;

#define NHEADS 4
#define BATCH  8
#define NN     1024
#define EMB    128
#define KD     32
#define FFH    512
#define LNELEM 131072.0f   // N*E per sample

// ---------------------------------------------------------------------------
// WMMA helpers (CDNA5 wave32, 16x16x32 f16 -> f32)
// ---------------------------------------------------------------------------
__device__ inline v8f wmma16(v16h a, v16h b, v8f c) {
    return __builtin_amdgcn_wmma_f32_16x16x32_f16(false, a, false, b, (short)0, c,
                                                  false, false);
}

// A-operand (16x32 f16) from row-major f16 rows (also valid as B-operand for
// Q*K^T since B[k,n] = Krow[n][k] uses the identical per-lane packing).
// Packing (ISA 7.12.2): lanes 0-15 -> row M=lane, halves K0..7 then K16..23;
// lanes 16-31 -> row M=lane-16, halves K8..15 then K24..31.
__device__ inline v16h frag_rows_f16(const _Float16* base, int stride) {
    int lane = threadIdx.x & 31;
    int row  = lane & 15;
    int o0   = (lane < 16) ? 0 : 8;
    const _Float16* p = base + row * stride + o0;
    v16h f;
    *(v8h*)&f     = *(const v8h*)(p);        // elements 0..7
    ((v8h*)&f)[1] = *(const v8h*)(p + 16);   // elements 8..15
    return f;
}

// Same packing but converting from f32 rows.
__device__ inline v16h frag_rows_f32(const float* base, int stride) {
    int lane = threadIdx.x & 31;
    int row  = lane & 15;
    int o0   = (lane < 16) ? 0 : 8;
    const float* p = base + row * stride + o0;
    v16h f;
#pragma unroll
    for (int j = 0; j < 8; ++j) {
        f[j]     = (_Float16)p[j];
        f[j + 8] = (_Float16)p[16 + j];
    }
    return f;
}

// f32 rows with layer-norm applied on the fly.
__device__ inline v16h frag_rows_f32_norm(const float* base, int stride,
                                          float mu, float rstd) {
    int lane = threadIdx.x & 31;
    int row  = lane & 15;
    int o0   = (lane < 16) ? 0 : 8;
    const float* p = base + row * stride + o0;
    v16h f;
#pragma unroll
    for (int j = 0; j < 8; ++j) {
        f[j]     = (_Float16)((p[j] - mu) * rstd);
        f[j + 8] = (_Float16)((p[16 + j] - mu) * rstd);
    }
    return f;
}

// B-operand from pre-packed fragment storage: tile t holds 32 lanes x 16
// contiguous halves (32 B/lane) -> two global_load_b128 per lane.
__device__ inline v16h frag_packedB(const _Float16* packed, int tile) {
    int lane = threadIdx.x & 31;
    const _Float16* p = packed + ((size_t)tile * 32 + lane) * 16;
    v16h f;
    *(v8h*)&f     = *(const v8h*)(p);
    ((v8h*)&f)[1] = *(const v8h*)(p + 8);
    return f;
}

// ---------------------------------------------------------------------------
// Pack kernels: swizzle a row-major [K][Nc] matrix into WMMA B-fragment
// layout (done ONCE per launch; every GEMM then loads contiguously).
// Tile t = kc * (Nc/16) + nc.  Lane halves j: j<8 -> kk=o0+j ; j>=8 ->
// kk=o0+16+(j-8), with o0 = (lane<16)?0:8, column = nc*16 + (lane&15).
// ---------------------------------------------------------------------------
__global__ void pack_b_f32_kernel(const float* __restrict__ src,
                                  _Float16* __restrict__ dst, int K, int Nc) {
    int idx = blockIdx.x * blockDim.x + threadIdx.x;   // (tile, lane)
    int total = (K >> 5) * (Nc >> 4) * 32;
    if (idx >= total) return;
    int lane = idx & 31, t = idx >> 5;
    int tilesPerK = Nc >> 4;
    int kc = t / tilesPerK, nc = t % tilesPerK;
    int col = nc * 16 + (lane & 15);
    int o0  = (lane < 16) ? 0 : 8;
    _Float16* d = dst + (size_t)idx * 16;
#pragma unroll
    for (int j = 0; j < 8; ++j) {
        d[j]     = (_Float16)src[(kc * 32 + o0 + j) * Nc + col];
        d[8 + j] = (_Float16)src[(kc * 32 + o0 + 16 + j) * Nc + col];
    }
}

__global__ void pack_b_f16_kernel(const _Float16* __restrict__ src,
                                  _Float16* __restrict__ dst, int K, int Nc) {
    int idx = blockIdx.x * blockDim.x + threadIdx.x;
    int total = (K >> 5) * (Nc >> 4) * 32;
    if (idx >= total) return;
    int lane = idx & 31, t = idx >> 5;
    int tilesPerK = Nc >> 4;
    int kc = t / tilesPerK, nc = t % tilesPerK;
    int col = nc * 16 + (lane & 15);
    int o0  = (lane < 16) ? 0 : 8;
    _Float16* d = dst + (size_t)idx * 16;
#pragma unroll
    for (int j = 0; j < 8; ++j) {
        d[j]     = src[(size_t)(kc * 32 + o0 + j) * Nc + col];
        d[8 + j] = src[(size_t)(kc * 32 + o0 + 16 + j) * Nc + col];
    }
}

// ---------------------------------------------------------------------------
// Kernel: QKV projections.  One wave per (h, b, 16-row tile).
// Q/K/V stored f16 as [H][B][N][KD] (row-major; V gets re-packed afterwards).
// ---------------------------------------------------------------------------
__global__ __launch_bounds__(256) void qkv_kernel(
    const float* __restrict__ x, const _Float16* __restrict__ Wqp,
    const _Float16* __restrict__ Wkp, const _Float16* __restrict__ Wvp,
    _Float16* __restrict__ Qb, _Float16* __restrict__ Kb,
    _Float16* __restrict__ Vb) {
    int wid  = (blockIdx.x * blockDim.x + threadIdx.x) >> 5;  // 0..2047
    int nt   = wid & 63;
    int b    = (wid >> 6) & 7;
    int h    = wid >> 9;
    int lane = threadIdx.x & 31;
    int n0   = nt * 16;
    int o0   = (lane < 16) ? 0 : 8;
    int colL = lane & 15;

    const float* xrow = x + (b * NN + n0) * EMB;
    v16h a[4];
#pragma unroll
    for (int kc = 0; kc < 4; ++kc) a[kc] = frag_rows_f32(xrow + kc * 32, EMB);

    const _Float16* Wp[3] = {Wqp, Wkp, Wvp};
    _Float16* Ob[3] = {Qb, Kb, Vb};
#pragma unroll
    for (int s = 0; s < 3; ++s) {
#pragma unroll
        for (int g = 0; g < 2; ++g) {  // output columns 16g..16g+15
            v8f c = {};
#pragma unroll
            for (int kc = 0; kc < 4; ++kc) {
                // tiles over [H*E][KD]: global kc = h*4+kc, tilesPerK = 2
                v16h bf = frag_packedB(Wp[s], (h * 4 + kc) * 2 + g);
                c = wmma16(a[kc], bf, c);
            }
#pragma unroll
            for (int r = 0; r < 8; ++r) {
                int row = n0 + r + o0;
                Ob[s][((h * BATCH + b) * NN + row) * KD + g * 16 + colL] =
                    (_Float16)c[r];
            }
        }
    }
}

// ---------------------------------------------------------------------------
// Kernel: fused attention.  Block = (b, 16-row tile), 256 threads (8 waves).
//   pass1: QK^T (WMMA) + input2 + 8->8->4 MLP (packed v_pk_fma_f16) -> S in LDS
//   pass2: softmax over m (store exp, keep 1/sum)
//   pass3: attn . V (WMMA, packed V), 2 waves per head
//   pass4: W_out projection (WMMA) + residual + LN1 partial sums
// ---------------------------------------------------------------------------
__global__ __launch_bounds__(256, 1) void attn_kernel(
    const _Float16* __restrict__ Qb, const _Float16* __restrict__ Kb,
    const _Float16* __restrict__ Vp, const float* __restrict__ in2,
    const float* __restrict__ Ws1, const float* __restrict__ bs1,
    const float* __restrict__ Ws2, const float* __restrict__ bs2,
    const _Float16* __restrict__ Wop, const float* __restrict__ x,
    float* __restrict__ res1, float2* __restrict__ part1) {
    __shared__ _Float16 S[NHEADS][16][NN];        // 128 KB score/prob tiles
    __shared__ float    partialV[NHEADS][16][KD]; // 8 KB
    __shared__ _Float16 headsLDS[NHEADS][16][KD]; // 4 KB
    __shared__ float    inv_sum[NHEADS][16];
    __shared__ _Float16 w1h[8][8], b1h[8], w2h[8][4], b2h[4];
    __shared__ float    redbuf[16];

    int tid = threadIdx.x, w = tid >> 5, lane = tid & 31;
    int nt = blockIdx.x & 63, b = blockIdx.x >> 6;
    int n0 = nt * 16;
    int o0 = (lane < 16) ? 0 : 8;
    int colL = lane & 15;

    if (tid < 64) w1h[tid >> 3][tid & 7] = (_Float16)Ws1[tid];
    if (tid < 8)  b1h[tid] = (_Float16)bs1[tid];
    if (tid < 32) w2h[tid >> 2][tid & 3] = (_Float16)Ws2[tid];
    if (tid < 4)  b2h[tid] = (_Float16)bs2[tid];
    __syncthreads();

    // Q fragments (16 rows x 32 K) per head
    v16h qf[NHEADS];
#pragma unroll
    for (int h = 0; h < NHEADS; ++h)
        qf[h] = frag_rows_f16(Qb + ((h * BATCH + b) * NN + n0) * KD, KD);

    // ---- pass 1: scores + head-mixing MLP -------------------------------
    for (int mt = w; mt < 64; mt += 8) {
        int m0 = mt * 16;
        // prefetch next tile's input2 rows (covers 4 heads x 16 rows in two
        // per-lane prefetch instructions -> global_prefetch_b8)
        if (mt + 8 < 64) {
            int m0n = m0 + 128;
            const float* pf0 =
                in2 + (size_t)(((lane >> 4) * BATCH + b) * NN + n0 + colL) * NN +
                m0n;
            const float* pf1 =
                in2 +
                (size_t)((((lane >> 4) + 2) * BATCH + b) * NN + n0 + colL) * NN +
                m0n;
            __builtin_prefetch(pf0, 0, 0);
            __builtin_prefetch(pf1, 0, 0);
        }
        v8f cf[NHEADS];
#pragma unroll
        for (int h = 0; h < NHEADS; ++h) {
            v16h kf = frag_rows_f16(Kb + ((h * BATCH + b) * NN + m0) * KD, KD);
            v8f z = {};
            cf[h] = wmma16(qf[h], kf, z);
        }
        float i2[NHEADS][8];
#pragma unroll
        for (int h = 0; h < NHEADS; ++h)
#pragma unroll
            for (int r = 0; r < 8; ++r)
                i2[h][r] = in2[((h * BATCH + b) * NN + (n0 + r + o0)) * NN +
                               m0 + colL];
        // MLP on h2 pairs (elements r0,r0+1) -> v_pk_fma_f16
#pragma unroll
        for (int rp = 0; rp < 4; ++rp) {
            h2 ch[8];
#pragma unroll
            for (int h = 0; h < NHEADS; ++h) {
                h2 t0 = {(_Float16)cf[h][2 * rp], (_Float16)cf[h][2 * rp + 1]};
                ch[h] = t0;
                h2 t1 = {(_Float16)i2[h][2 * rp], (_Float16)i2[h][2 * rp + 1]};
                ch[4 + h] = t1;
            }
            h2 y[8];
#pragma unroll
            for (int j = 0; j < 8; ++j) {
                h2 acc = {b1h[j], b1h[j]};
#pragma unroll
                for (int i = 0; i < 8; ++i) {
                    h2 wv = {w1h[i][j], w1h[i][j]};
                    acc += ch[i] * wv;
                }
                h2 zero = {};
                y[j] = __builtin_elementwise_max(acc, zero);
            }
#pragma unroll
            for (int k = 0; k < 4; ++k) {
                h2 acc = {b2h[k], b2h[k]};
#pragma unroll
                for (int j = 0; j < 8; ++j) {
                    h2 wv = {w2h[j][k], w2h[j][k]};
                    acc += y[j] * wv;
                }
                S[k][2 * rp + o0][m0 + colL]     = acc[0];
                S[k][2 * rp + 1 + o0][m0 + colL] = acc[1];
            }
        }
    }
    __syncthreads();

    // ---- pass 2: softmax over m (64 (h,row) pairs, 8 per wave) ----------
    for (int rr = w * 8; rr < w * 8 + 8; ++rr) {
        int h = rr >> 4, row = rr & 15;
        _Float16* srow = &S[h][row][0];
        float mx = -3.4e38f;
#pragma unroll 4
        for (int t = 0; t < 32; ++t)
            mx = fmaxf(mx, (float)srow[lane + 32 * t]);
#pragma unroll
        for (int off = 16; off > 0; off >>= 1)
            mx = fmaxf(mx, __shfl_xor(mx, off, 32));
        float sum = 0.0f;
#pragma unroll 4
        for (int t = 0; t < 32; ++t) {
            float p = __expf((float)srow[lane + 32 * t] - mx);
            srow[lane + 32 * t] = (_Float16)p;
            sum += p;
        }
#pragma unroll
        for (int off = 16; off > 0; off >>= 1)
            sum += __shfl_xor(sum, off, 32);
        if (lane == 0) inv_sum[h][row] = 1.0f / sum;
    }
    __syncthreads();

    // ---- pass 3: attn . V  (waves h and h+4 split the 32 m-chunks) ------
    {
        int h = w & 3, hv = w >> 2;
        int tbase = ((h * BATCH + b) * (NN / 32)) * 2;  // packed-V tile base
        v8f cv0 = {}, cv1 = {};
        for (int mc = hv * 16; mc < hv * 16 + 16; ++mc) {
            int m0 = mc * 32;
            v16h af = frag_rows_f16(&S[h][0][m0], NN);
            v16h b0 = frag_packedB(Vp, tbase + mc * 2);
            v16h b1 = frag_packedB(Vp, tbase + mc * 2 + 1);
            cv0 = wmma16(af, b0, cv0);
            cv1 = wmma16(af, b1, cv1);
        }
        if (hv == 1) {
#pragma unroll
            for (int r = 0; r < 8; ++r) {
                partialV[h][r + o0][colL]      = cv0[r];
                partialV[h][r + o0][16 + colL] = cv1[r];
            }
        }
        __syncthreads();
        if (hv == 0) {
#pragma unroll
            for (int r = 0; r < 8; ++r) {
                int row = r + o0;
                float sc = inv_sum[h][row];
                headsLDS[h][row][colL] =
                    (_Float16)((cv0[r] + partialV[h][row][colL]) * sc);
                headsLDS[h][row][16 + colL] =
                    (_Float16)((cv1[r] + partialV[h][row][16 + colL]) * sc);
            }
        }
        __syncthreads();
    }

    // ---- pass 4: W_out projection + residual + LN1 partials -------------
    v8f co = {};
#pragma unroll
    for (int h = 0; h < NHEADS; ++h) {
        v16h af = frag_rows_f16(&headsLDS[h][0][0], KD);
        // Wop tiles over [H*KD=128][EMB=128]: kc = h, tilesPerK = 8
        v16h bf = frag_packedB(Wop, h * 8 + w);
        co = wmma16(af, bf, co);
    }
    float s1 = 0.0f, s2 = 0.0f;
#pragma unroll
    for (int r = 0; r < 8; ++r) {
        int row = n0 + r + o0;
        int col = w * 16 + colL;
        float v = co[r] + x[(b * NN + row) * EMB + col];
        res1[(b * NN + row) * EMB + col] = v;
        s1 += v;
        s2 += v * v;
    }
#pragma unroll
    for (int off = 16; off > 0; off >>= 1) {
        s1 += __shfl_xor(s1, off, 32);
        s2 += __shfl_xor(s2, off, 32);
    }
    if (lane == 0) { redbuf[w * 2] = s1; redbuf[w * 2 + 1] = s2; }
    __syncthreads();
    if (tid == 0) {
        float a = 0.0f, c2 = 0.0f;
#pragma unroll
        for (int i = 0; i < 8; ++i) { a += redbuf[2 * i]; c2 += redbuf[2 * i + 1]; }
        part1[blockIdx.x] = make_float2(a, c2);
    }
}

// ---------------------------------------------------------------------------
// Kernel: combine per-tile LN partials -> (mean, rstd) per sample (fixed order)
// ---------------------------------------------------------------------------
__global__ void stats_kernel(const float2* __restrict__ part,
                             float2* __restrict__ stats) {
    int b = blockIdx.x;
    if (threadIdx.x == 0) {
        float s1 = 0.0f, s2 = 0.0f;
        for (int i = 0; i < 64; ++i) {
            float2 p = part[b * 64 + i];
            s1 += p.x;
            s2 += p.y;
        }
        float mean = s1 / LNELEM;
        float var  = (s2 - LNELEM * mean * mean) / (LNELEM - 1.0f);  // ddof=1
        stats[b] = make_float2(mean, rsqrtf(var + 1e-5f));
    }
}

// ---------------------------------------------------------------------------
// Kernel: FF1 = relu( LN1(res1) @ W_ff1 ).  Block = 16-row tile; wave = 64 cols.
// ---------------------------------------------------------------------------
__global__ __launch_bounds__(256) void ff1_kernel(
    const float* __restrict__ res1, const float2* __restrict__ stats1,
    const _Float16* __restrict__ Wff1p, _Float16* __restrict__ hff) {
    int rt = blockIdx.x;          // 0..511 over B*N/16 rows
    int b = rt >> 6;
    int w = threadIdx.x >> 5, lane = threadIdx.x & 31;
    int o0 = (lane < 16) ? 0 : 8, colL = lane & 15;
    float mu = stats1[b].x, rstd = stats1[b].y;

    v16h a[4];
#pragma unroll
    for (int kc = 0; kc < 4; ++kc)
        a[kc] = frag_rows_f32_norm(res1 + rt * 16 * EMB + kc * 32, EMB, mu, rstd);

#pragma unroll
    for (int ct = 0; ct < 4; ++ct) {
        int c0 = w * 64 + ct * 16;
        v8f c = {};
#pragma unroll
        for (int kc = 0; kc < 4; ++kc) {
            // Wff1 tiles over [128][512]: tilesPerK = 32
            v16h bf = frag_packedB(Wff1p, kc * 32 + w * 4 + ct);
            c = wmma16(a[kc], bf, c);
        }
#pragma unroll
        for (int r = 0; r < 8; ++r)
            hff[(rt * 16 + r + o0) * FFH + c0 + colL] =
                (_Float16)fmaxf(c[r], 0.0f);
    }
}

// ---------------------------------------------------------------------------
// Kernel: FF2 = hff @ W_ff2 + LN1(res1); writes res2 + LN2 partials.
// ---------------------------------------------------------------------------
__global__ __launch_bounds__(256) void ff2_kernel(
    const _Float16* __restrict__ hff, const _Float16* __restrict__ Wff2p,
    const float* __restrict__ res1, const float2* __restrict__ stats1,
    float* __restrict__ res2, float2* __restrict__ part2) {
    __shared__ float redbuf[16];
    int rt = blockIdx.x;
    int b = rt >> 6;
    int tid = threadIdx.x, w = tid >> 5, lane = tid & 31;
    int o0 = (lane < 16) ? 0 : 8, colL = lane & 15;

    v8f c = {};
    for (int kc = 0; kc < 16; ++kc) {
        v16h a  = frag_rows_f16(hff + rt * 16 * FFH + kc * 32, FFH);
        // Wff2 tiles over [512][128]: tilesPerK = 8
        v16h bf = frag_packedB(Wff2p, kc * 8 + w);
        c = wmma16(a, bf, c);
    }
    float mu = stats1[b].x, rstd = stats1[b].y;
    float s1 = 0.0f, s2 = 0.0f;
#pragma unroll
    for (int r = 0; r < 8; ++r) {
        int row = rt * 16 + r + o0;
        int col = w * 16 + colL;
        float x1 = (res1[row * EMB + col] - mu) * rstd;
        float v  = c[r] + x1;
        res2[row * EMB + col] = v;
        s1 += v;
        s2 += v * v;
    }
#pragma unroll
    for (int off = 16; off > 0; off >>= 1) {
        s1 += __shfl_xor(s1, off, 32);
        s2 += __shfl_xor(s2, off, 32);
    }
    if (lane == 0) { redbuf[w * 2] = s1; redbuf[w * 2 + 1] = s2; }
    __syncthreads();
    if (tid == 0) {
        float a = 0.0f, c2 = 0.0f;
#pragma unroll
        for (int i = 0; i < 8; ++i) { a += redbuf[2 * i]; c2 += redbuf[2 * i + 1]; }
        part2[blockIdx.x] = make_float2(a, c2);
    }
}

// ---------------------------------------------------------------------------
// Kernel: final LN2 normalize -> d_out[0 : B*N*E]
// ---------------------------------------------------------------------------
__global__ void norm_kernel(const float* __restrict__ res2,
                            const float2* __restrict__ stats2,
                            float* __restrict__ out) {
    int i = blockIdx.x * blockDim.x + threadIdx.x;  // over B*N*E = 1M
    int b = i >> 17;                                 // N*E = 131072
    out[i] = (res2[i] - stats2[b].x) * stats2[b].y;
}

// ---------------------------------------------------------------------------
// Host launcher
// ---------------------------------------------------------------------------
extern "C" void kernel_launch(void* const* d_in, const int* in_sizes, int n_in,
                              void* d_out, int out_size, void* d_ws,
                              size_t ws_size, hipStream_t stream) {
    const float* input1 = (const float*)d_in[0];
    const float* input2 = (const float*)d_in[1];
    const float* Wq = (const float*)d_in[2];
    const float* Wk = (const float*)d_in[3];
    const float* Wv = (const float*)d_in[4];
    const float* Wo = (const float*)d_in[5];
    const float* Ws1 = (const float*)d_in[6];
    const float* bs1 = (const float*)d_in[7];
    const float* Ws2 = (const float*)d_in[8];
    const float* bs2 = (const float*)d_in[9];
    const float* Wff1 = (const float*)d_in[10];
    const float* Wff2 = (const float*)d_in[11];
    float* out = (float*)d_out;

    size_t off = 0;
    auto wsp = [&](size_t bytes) {
        char* p = (char*)d_ws + off;
        off += (bytes + 255) & ~(size_t)255;
        return (void*)p;
    };
    _Float16* Qb   = (_Float16*)wsp((size_t)NHEADS * BATCH * NN * KD * 2);
    _Float16* Kb   = (_Float16*)wsp((size_t)NHEADS * BATCH * NN * KD * 2);
    _Float16* Vb   = (_Float16*)wsp((size_t)NHEADS * BATCH * NN * KD * 2);
    _Float16* Vp   = (_Float16*)wsp((size_t)NHEADS * BATCH * NN * KD * 2);
    _Float16* Wqp  = (_Float16*)wsp((size_t)NHEADS * EMB * KD * 2);
    _Float16* Wkp  = (_Float16*)wsp((size_t)NHEADS * EMB * KD * 2);
    _Float16* Wvp  = (_Float16*)wsp((size_t)NHEADS * EMB * KD * 2);
    _Float16* Wop  = (_Float16*)wsp((size_t)NHEADS * KD * EMB * 2);
    _Float16* Wff1p = (_Float16*)wsp((size_t)EMB * FFH * 2);
    _Float16* Wff2p = (_Float16*)wsp((size_t)FFH * EMB * 2);
    float*  res1  = (float*)wsp((size_t)BATCH * NN * EMB * 4);
    _Float16* hff = (_Float16*)wsp((size_t)BATCH * NN * FFH * 2);
    float*  res2  = (float*)wsp((size_t)BATCH * NN * EMB * 4);
    float2* part1 = (float2*)wsp(512 * sizeof(float2));
    float2* part2 = (float2*)wsp(512 * sizeof(float2));
    float2* stats1 = (float2*)wsp(BATCH * sizeof(float2));
    float2* stats2 = (float2*)wsp(BATCH * sizeof(float2));

    // weights -> f16, pre-swizzled into WMMA B-fragment layout (one-time)
    auto packW = [&](const float* s, _Float16* d, int K, int Nc) {
        int items = (K >> 5) * (Nc >> 4) * 32;
        pack_b_f32_kernel<<<(items + 255) / 256, 256, 0, stream>>>(s, d, K, Nc);
    };
    packW(Wq, Wqp, NHEADS * EMB, KD);   // [512][32]
    packW(Wk, Wkp, NHEADS * EMB, KD);
    packW(Wv, Wvp, NHEADS * EMB, KD);
    packW(Wo, Wop, NHEADS * KD, EMB);   // [128][128]
    packW(Wff1, Wff1p, EMB, FFH);       // [128][512]
    packW(Wff2, Wff2p, FFH, EMB);       // [512][128]

    // QKV projections: 2048 waves
    qkv_kernel<<<256, 256, 0, stream>>>(input1, Wqp, Wkp, Wvp, Qb, Kb, Vb);

    // re-pack V into B-fragment layout (read once, 2 MB)
    {
        int items = ((NHEADS * BATCH * NN) >> 5) * (KD >> 4) * 32;
        pack_b_f16_kernel<<<(items + 255) / 256, 256, 0, stream>>>(
            Vb, Vp, NHEADS * BATCH * NN, KD);
    }

    // fused attention (512 blocks = B * N/16)
    attn_kernel<<<512, 256, 0, stream>>>(Qb, Kb, Vp, input2, Ws1, bs1, Ws2,
                                         bs2, Wop, input1, res1, part1);
    stats_kernel<<<BATCH, 32, 0, stream>>>(part1, stats1);

    ff1_kernel<<<512, 256, 0, stream>>>(res1, stats1, Wff1p, hff);
    ff2_kernel<<<512, 256, 0, stream>>>(hff, Wff2p, res1, stats1, res2, part2);
    stats_kernel<<<BATCH, 32, 0, stream>>>(part2, stats2);

    norm_kernel<<<4096, 256, 0, stream>>>(res2, stats2, out);

    // second tuple element: pass input2 through
    hipMemcpyAsync(out + (size_t)BATCH * NN * EMB, input2,
                   (size_t)NHEADS * BATCH * NN * NN * sizeof(float),
                   hipMemcpyDeviceToDevice, stream);
}
